// MolEncoder_6734508720420
// MI455X (gfx1250) — matrix operand order repaired and checked by reading, not compiled, not run on hardware
//
#include <hip/hip_runtime.h>

#define HDIM  128
#define MAXNB 6
#define MTILE 16     // edges per block (M of the GEMM tile)
#define KT1   8      // 256 / 32 k-steps for layer 1
#define KT2   4      // 128 / 32 k-steps for layer 2
#define NT    8      // 128 / 16 n-tiles (one wave each)

typedef __attribute__((ext_vector_type(16))) __bf16 v16bf;
typedef __attribute__((ext_vector_type(8)))  __bf16 v8bf;
typedef __attribute__((ext_vector_type(4)))  __bf16 v4bf;
typedef __attribute__((ext_vector_type(8)))  float  v8f;
typedef __attribute__((ext_vector_type(4)))  float  v4f;

// ---------------------------------------------------------------------------
// Kernel 1: hnode[n,:] = sum_j hatom[dgraph[n,j],:]   (f32 accumulate -> bf16)
// One wave32 per node row: each lane owns one float4 (128 floats / 32 lanes).
// ---------------------------------------------------------------------------
__global__ void __launch_bounds__(256)
hnode_gather_sum(const float* __restrict__ hatom,
                 const int*   __restrict__ dgraph,
                 unsigned short* __restrict__ hnode_us,
                 int n_nodes)
{
    __bf16* hnode = reinterpret_cast<__bf16*>(hnode_us);
    const int wave = (int)((blockIdx.x * blockDim.x + threadIdx.x) >> 5);
    const int lane = threadIdx.x & 31;
    if (wave >= n_nodes) return;

    v4f acc = {0.f, 0.f, 0.f, 0.f};
#pragma unroll
    for (int j = 0; j < MAXNB; ++j) {
        const int idx = dgraph[(size_t)wave * MAXNB + j];
        const v4f* row = (const v4f*)(hatom + (size_t)idx * HDIM);
        acc += row[lane];
    }
    v4bf o;
    o.x = (__bf16)acc.x; o.y = (__bf16)acc.y;
    o.z = (__bf16)acc.z; o.w = (__bf16)acc.w;
    *(v4bf*)(hnode + (size_t)wave * HDIM + lane * 4) = o;
}

// ---------------------------------------------------------------------------
// Kernel 2: pack Wg1 [256x128] and Wg2 [128x128] (f32 row-major, x@W form)
// into bf16 B-fragment layout for v_wmma_f32_16x16x32_bf16:
//   lane l holds column N = nt*16 + (l&15);
//   its 16 halves are K = kt*32 + ((l&16)?8:0) + (h<8 ? h : h+8).
// Packed element index: ((nt*KT + kt)*32 + lane)*16 + h  -> each lane reads
// one contiguous 32-byte chunk per k-step in the GEMM.
// ---------------------------------------------------------------------------
__global__ void __launch_bounds__(256)
pack_weights(const float* __restrict__ Wg1, const float* __restrict__ Wg2,
             unsigned short* __restrict__ w1p_us,
             unsigned short* __restrict__ w2p_us)
{
    __bf16* w1p = reinterpret_cast<__bf16*>(w1p_us);
    __bf16* w2p = reinterpret_cast<__bf16*>(w2p_us);
    const int W1N = NT * KT1 * 32 * 16;   // 32768
    const int W2N = NT * KT2 * 32 * 16;   // 16384
    const int t = blockIdx.x * blockDim.x + threadIdx.x;

    if (t < W1N) {
        const int h = t & 15, lane = (t >> 4) & 31, tile = t >> 9;
        const int kt = tile % KT1, nt = tile / KT1;
        const int n = nt * 16 + (lane & 15);
        const int k = kt * 32 + ((lane & 16) ? 8 : 0) + (h < 8 ? h : h + 8);
        w1p[t] = (__bf16)Wg1[k * HDIM + n];
    } else if (t < W1N + W2N) {
        const int u = t - W1N;
        const int h = u & 15, lane = (u >> 4) & 31, tile = u >> 9;
        const int kt = tile % KT2, nt = tile / KT2;
        const int n = nt * 16 + (lane & 15);
        const int k = kt * 32 + ((lane & 16) ? 8 : 0) + (h < 8 ? h : h + 8);
        w2p[u] = (__bf16)Wg2[k * HDIM + n];
    }
}

// ---------------------------------------------------------------------------
// Kernel 3 (fused): per block of 16 edges
//   x = [ hnode_bf16[fmess_src[e]]  |  bf16( sum_j hatom[cgraph[e,j]] ) ]
//   h1 = relu(x @ W1 + b1)   (WMMA bf16, f32 acc, 8 k-steps)
//   out = h1 @ W2 + b2       (WMMA bf16, f32 acc, 4 k-steps)
// 256 threads = 8 waves; wave w computes the 16-wide N-tile w.
// ---------------------------------------------------------------------------
__global__ void __launch_bounds__(256)
mpnn_fused(const float* __restrict__ hatom,
           const int*   __restrict__ cgraph,
           const int*   __restrict__ fmess_src,
           const unsigned short* __restrict__ hnode_us,
           const unsigned short* __restrict__ w1p_us,
           const unsigned short* __restrict__ w2p_us,
           const float* __restrict__ bg1,
           const float* __restrict__ bg2,
           float* __restrict__ out,
           int n_edges)
{
    const __bf16* hnode = reinterpret_cast<const __bf16*>(hnode_us);
    const __bf16* w1p   = reinterpret_cast<const __bf16*>(w1p_us);
    const __bf16* w2p   = reinterpret_cast<const __bf16*>(w2p_us);

    __shared__ __bf16 xbuf[MTILE][2 * HDIM];   // 16 x 256 bf16 = 8 KB
    __shared__ __bf16 h1buf[MTILE][HDIM];      // 16 x 128 bf16 = 4 KB

    const int tid  = threadIdx.x;
    const int lane = tid & 31;
    const int nt   = tid >> 5;                 // wave id == N-tile index
    const long long e0 = (long long)blockIdx.x * MTILE;

    // ---- Stage X: build x = [hmess1 | hmess2] in LDS -----------------------
    {
        const int m   = tid >> 4;              // edge row within tile, 0..15
        const int seg = tid & 15;              // 8-half (16 B) segment
        long long e = e0 + m;
        if (e > (long long)n_edges - 1) e = (long long)n_edges - 1;

        // hmess1: bf16 gather from precomputed hnode
        const int src = fmess_src[e];
        v8bf v1 = *(const v8bf*)(hnode + (size_t)src * HDIM + seg * 8);
        *(v8bf*)(&xbuf[m][seg * 8]) = v1;

        // hmess2: f32 gather-sum over 6 neighbors, then cvt to bf16
        float acc[8];
#pragma unroll
        for (int i = 0; i < 8; ++i) acc[i] = 0.f;
#pragma unroll
        for (int j = 0; j < MAXNB; ++j) {
            const int idx = cgraph[e * MAXNB + j];
            const float* p = hatom + (size_t)idx * HDIM + seg * 8;
            v4f a0 = *(const v4f*)(p);
            v4f a1 = *(const v4f*)(p + 4);
            acc[0] += a0.x; acc[1] += a0.y; acc[2] += a0.z; acc[3] += a0.w;
            acc[4] += a1.x; acc[5] += a1.y; acc[6] += a1.z; acc[7] += a1.w;
        }
        v8bf o;
#pragma unroll
        for (int i = 0; i < 8; ++i) o[i] = (__bf16)acc[i];
        *(v8bf*)(&xbuf[m][HDIM + seg * 8]) = o;
    }
    __syncthreads();

    // A-fragment addressing (16-bit A 16x32 layout):
    //   lane -> row M = lane&15 ; halves = K {0-7,16-23} or {8-15,24-31}
    const int am   = lane & 15;
    const int ksel = (lane & 16) ? 8 : 0;
    const int ncol = nt * 16 + am;             // C/D column for this lane

    // ---- GEMM1: relu(x @ W1 + b1) -> h1buf (bf16) ---------------------------
    {
        v8f acc = {};
#pragma unroll
        for (int kt = 0; kt < KT1; ++kt) {
            union { v16bf v; v8bf h[2]; } a, b;
            const __bf16* ap = &xbuf[am][kt * 32 + ksel];
            a.h[0] = *(const v8bf*)(ap);            // K base+0 .. +7
            a.h[1] = *(const v8bf*)(ap + 16);       // K base+16 .. +23
            b.v = *(const v16bf*)(w1p + ((size_t)(nt * KT1 + kt) * 32 + lane) * 16);
            acc = __builtin_amdgcn_wmma_f32_16x16x32_bf16(
                      false, a.v, false, b.v, (short)0, acc, false, false);
        }
        const float bias = bg1[ncol];
#pragma unroll
        for (int v = 0; v < 8; ++v) {
            float r = acc[v] + bias;
            r = r > 0.f ? r : 0.f;
            h1buf[v + ksel][ncol] = (__bf16)r;      // D: M = v + 8*(lane>=16)
        }
    }
    __syncthreads();

    // ---- GEMM2: h1 @ W2 + b2 -> out (f32) -----------------------------------
    {
        v8f acc = {};
#pragma unroll
        for (int kt = 0; kt < KT2; ++kt) {
            union { v16bf v; v8bf h[2]; } a, b;
            const __bf16* ap = &h1buf[am][kt * 32 + ksel];
            a.h[0] = *(const v8bf*)(ap);
            a.h[1] = *(const v8bf*)(ap + 16);
            b.v = *(const v16bf*)(w2p + ((size_t)(nt * KT2 + kt) * 32 + lane) * 16);
            acc = __builtin_amdgcn_wmma_f32_16x16x32_bf16(
                      false, a.v, false, b.v, (short)0, acc, false, false);
        }
        const float bias = bg2[ncol];
#pragma unroll
        for (int v = 0; v < 8; ++v) {
            const long long row = e0 + v + ksel;
            if (row < n_edges)
                out[row * HDIM + ncol] = acc[v] + bias;
        }
    }
}

// ---------------------------------------------------------------------------
extern "C" void kernel_launch(void* const* d_in, const int* in_sizes, int n_in,
                              void* d_out, int out_size, void* d_ws, size_t ws_size,
                              hipStream_t stream)
{
    const float* hatom = (const float*)d_in[0];
    const int*   dgraph = (const int*)d_in[1];
    const int*   cgraph = (const int*)d_in[2];
    const int*   fmess  = (const int*)d_in[3];
    const float* Wg1    = (const float*)d_in[4];
    const float* bg1    = (const float*)d_in[5];
    const float* Wg2    = (const float*)d_in[6];
    const float* bg2    = (const float*)d_in[7];
    float* out = (float*)d_out;

    const int n_nodes = in_sizes[1] / MAXNB;   // 150000
    const int n_edges = in_sizes[2] / MAXNB;   // 300000

    // Workspace layout: [W1 packed 64KB][W2 packed 32KB + pad][hnode bf16]
    char* ws = (char*)d_ws;
    unsigned short* w1p   = (unsigned short*)ws;
    unsigned short* w2p   = (unsigned short*)(ws + (64 << 10));
    unsigned short* hnode = (unsigned short*)(ws + (128 << 10));

    const int W1N = NT * KT1 * 32 * 16;
    const int W2N = NT * KT2 * 32 * 16;

    pack_weights<<<(W1N + W2N + 255) / 256, 256, 0, stream>>>(Wg1, Wg2, w1p, w2p);
    hnode_gather_sum<<<(n_nodes + 7) / 8, 256, 0, stream>>>(hatom, dgraph, hnode, n_nodes);
    mpnn_fused<<<(n_edges + MTILE - 1) / MTILE, 256, 0, stream>>>(
        hatom, cgraph, fmess, hnode, w1p, w2p, bg1, bg2, out, n_edges);
}
